// SGCALite_43009802502674
// MI455X (gfx1250) — compile-verified, matrix operand order checked by reading, not run on hardware
//
#include <hip/hip_runtime.h>
#include <hip/hip_bf16.h>

typedef _Float16 half_t;
typedef __attribute__((ext_vector_type(16))) _Float16 v16h;
typedef __attribute__((ext_vector_type(8)))  _Float16 v8h;
typedef __attribute__((ext_vector_type(8)))  float    v8f;

#define CLOUD_L 32768
#define NPTS    131072   // B*L
#define NCLOUD  4
#define DIM     256

static __device__ __forceinline__ v16h ldfrag(const half_t* p0, const half_t* p1) {
    v8h lo = *(const v8h*)p0;
    v8h hi = *(const v8h*)p1;
    v16h r;
#pragma unroll
    for (int i = 0; i < 8; ++i) { r[i] = lo[i]; r[i + 8] = hi[i]; }
    return r;
}

static __device__ __forceinline__ v8f wmma16(const v16h& a, const v16h& b, const v8f& c) {
    return __builtin_amdgcn_wmma_f32_16x16x32_f16(false, a, false, b, (short)0, c, false, false);
}

// ---------------------------------------------------------------------------
// Weight transpose + f16 convert: Wt[n*K + k] = (half) W[(k+rowOff)*N + n]
// ---------------------------------------------------------------------------
__global__ __launch_bounds__(256) void k_transpose(const float* __restrict__ W,
                                                   half_t* __restrict__ Wt,
                                                   int K, int N, int rowOff) {
    int i = blockIdx.x * 256 + threadIdx.x;
    if (i < K * N) {
        int n = i / K, k = i - n * K;
        Wt[i] = (half_t)W[(size_t)(k + rowOff) * N + n];
    }
}

// ---------------------------------------------------------------------------
// Per-cloud coordinate stats: mean, max|c-mean|, min, 1/range per dim
// stats[c*12 + {0..2 mean, 3..5 scale, 6..8 cmin, 9..11 invrange}]
// ---------------------------------------------------------------------------
__global__ __launch_bounds__(1024) void k_reduce(const float* __restrict__ coord,
                                                 float* __restrict__ stats) {
    int c = blockIdx.x, tid = threadIdx.x;
    __shared__ float red[1024];
    float sm[3] = {0.f, 0.f, 0.f};
    float mn[3] = {3.4e38f, 3.4e38f, 3.4e38f};
    float mx[3] = {-3.4e38f, -3.4e38f, -3.4e38f};
    for (int i = tid; i < CLOUD_L; i += 1024) {
        const float* p = coord + ((size_t)c * CLOUD_L + i) * 3;
#pragma unroll
        for (int d = 0; d < 3; ++d) {
            float v = p[d];
            sm[d] += v; mn[d] = fminf(mn[d], v); mx[d] = fmaxf(mx[d], v);
        }
    }
    float res[9];
    for (int q = 0; q < 9; ++q) {
        float v = (q < 3) ? sm[q] : (q < 6 ? mn[q - 3] : mx[q - 6]);
        red[tid] = v;
        __syncthreads();
        for (int s = 512; s > 0; s >>= 1) {
            if (tid < s) {
                float a = red[tid], b = red[tid + s];
                red[tid] = (q < 3) ? (a + b) : (q < 6 ? fminf(a, b) : fmaxf(a, b));
            }
            __syncthreads();
        }
        res[q] = red[0];
        __syncthreads();
    }
    if (tid == 0) {
#pragma unroll
        for (int d = 0; d < 3; ++d) {
            float mean = res[d] / (float)CLOUD_L;
            float lo = res[3 + d], hi = res[6 + d];
            stats[c * 12 + d]     = mean;
            stats[c * 12 + 3 + d] = fmaxf(fmaxf(hi - mean, mean - lo), 1e-6f);
            stats[c * 12 + 6 + d] = lo;
            stats[c * 12 + 9 + d] = 1.f / fmaxf(hi - lo, 1e-6f);
        }
    }
}

// ---------------------------------------------------------------------------
// Per-point: PE MLP, x = LN(f+pe, pre), h1 = LN(x, n1), sort key.
// One wave (32 lanes) per point, 8 dims per lane.
// ---------------------------------------------------------------------------
__global__ __launch_bounds__(256) void k_point(
    const float* __restrict__ feats, const float* __restrict__ coord,
    const float* __restrict__ stats,
    const float* __restrict__ pe_w1, const float* __restrict__ pe_b1,
    const float* __restrict__ pe_w2, const float* __restrict__ pe_b2,
    const float* __restrict__ pre_g, const float* __restrict__ pre_b,
    const float* __restrict__ n1_g, const float* __restrict__ n1_b,
    half_t* __restrict__ x16, half_t* __restrict__ h1,
    float* __restrict__ keys) {
    int tid = threadIdx.x, wave = tid >> 5, lane = tid & 31;
    int pt = blockIdx.x * 8 + wave;
    int cloud = pt / CLOUD_L;
    const float* st = stats + cloud * 12;
    __shared__ float hsh[8][64];
    float cd[3];
#pragma unroll
    for (int d = 0; d < 3; ++d) cd[d] = coord[(size_t)pt * 3 + d];
    for (int j = lane; j < 64; j += 32) {
        float a = pe_b1[j];
#pragma unroll
        for (int d = 0; d < 3; ++d)
            a += ((cd[d] - st[d]) / st[3 + d]) * pe_w1[d * 64 + j];
        hsh[wave][j] = fmaxf(a, 0.f);
    }
    __syncthreads();
    int o0 = lane * 8;
    float pe[8];
#pragma unroll
    for (int i = 0; i < 8; ++i) pe[i] = pe_b2[o0 + i];
    for (int j = 0; j < 64; ++j) {
        float hj = hsh[wave][j];
        const float* w = pe_w2 + j * 256 + o0;
#pragma unroll
        for (int i = 0; i < 8; ++i) pe[i] += hj * w[i];
    }
    float xv[8];
#pragma unroll
    for (int i = 0; i < 8; ++i)
        xv[i] = feats[(size_t)pt * DIM + o0 + i] + pe[i];
    // LN with pre_g/pre_b
    float s = 0.f, s2 = 0.f;
#pragma unroll
    for (int i = 0; i < 8; ++i) { s += xv[i]; s2 += xv[i] * xv[i]; }
#pragma unroll
    for (int o = 16; o > 0; o >>= 1) { s += __shfl_xor(s, o); s2 += __shfl_xor(s2, o); }
    float mean = s * (1.f / DIM);
    float inv = rsqrtf(s2 * (1.f / DIM) - mean * mean + 1e-5f);
    float x1[8];
#pragma unroll
    for (int i = 0; i < 8; ++i) {
        x1[i] = (xv[i] - mean) * inv * pre_g[o0 + i] + pre_b[o0 + i];
        x16[(size_t)pt * DIM + o0 + i] = (half_t)x1[i];
    }
    // LN with n1_g/n1_b
    s = 0.f; s2 = 0.f;
#pragma unroll
    for (int i = 0; i < 8; ++i) { s += x1[i]; s2 += x1[i] * x1[i]; }
#pragma unroll
    for (int o = 16; o > 0; o >>= 1) { s += __shfl_xor(s, o); s2 += __shfl_xor(s2, o); }
    float mean2 = s * (1.f / DIM);
    float inv2 = rsqrtf(s2 * (1.f / DIM) - mean2 * mean2 + 1e-5f);
#pragma unroll
    for (int i = 0; i < 8; ++i)
        h1[(size_t)pt * DIM + o0 + i] =
            (half_t)((x1[i] - mean2) * inv2 * n1_g[o0 + i] + n1_b[o0 + i]);
    if (lane == 0) {
        float k0 = (cd[0] - st[6]) * st[9];
        float k1 = (cd[1] - st[7]) * st[10];
        float k2 = (cd[2] - st[8]) * st[11];
        keys[pt] = k0 + 2.17f * k1 + 3.31f * k2;
    }
}

// ---------------------------------------------------------------------------
// Per-cloud bitonic argsort over keys (one block/cloud, data in global ws).
// order[c*L + s] = c*L + local_index_of_sth_smallest_key
// ---------------------------------------------------------------------------
__global__ __launch_bounds__(1024) void k_sort(const float* __restrict__ keys,
                                               float* __restrict__ kbuf,
                                               int* __restrict__ ibuf,
                                               int* __restrict__ order) {
    int c = blockIdx.x;
    float* kk = kbuf + (size_t)c * CLOUD_L;
    int* ii = ibuf + (size_t)c * CLOUD_L;
    for (int i = threadIdx.x; i < CLOUD_L; i += 1024) {
        kk[i] = keys[(size_t)c * CLOUD_L + i];
        ii[i] = i;
    }
    __syncthreads();
    for (int size = 2; size <= CLOUD_L; size <<= 1) {
        for (int stride = size >> 1; stride > 0; stride >>= 1) {
            for (int t = threadIdx.x; t < CLOUD_L / 2; t += 1024) {
                int i = 2 * t - (t & (stride - 1));
                int j = i + stride;
                bool up = ((i & size) == 0);
                float a = kk[i], b = kk[j];
                bool sw = up ? (a > b) : (a < b);
                if (sw) {
                    kk[i] = b; kk[j] = a;
                    int ti = ii[i]; ii[i] = ii[j]; ii[j] = ti;
                }
            }
            __syncthreads();
        }
    }
    for (int i = threadIdx.x; i < CLOUD_L; i += 1024)
        order[(size_t)c * CLOUD_L + i] = c * CLOUD_L + ii[i];
}

// ---------------------------------------------------------------------------
// Generic WMMA GEMM: C[32 x ND] per block = act(A[32 x KD] @ Wt^T + bias) ...
// A row-major f16 (or f32 via A32), Wt is [ND][KD] f16. 8 waves, 256 threads.
// MODE: 0 f16 store, 1 relu f16, 2 gelu f16, 3 sigmoid f32,
//       4 +res -> xs2 f16 + LN -> h2 f16, 5 +res -> scatter f16 + gsum,
//       6 out = fin + gate * v (f32)
// ---------------------------------------------------------------------------
template <int KD, int ND, int MODE>
__global__ __launch_bounds__(256) void gemm_k(
    const half_t* __restrict__ Ah, const float* __restrict__ A32,
    const int* __restrict__ rowmap,
    const half_t* __restrict__ Wt,
    const float* __restrict__ bias, int biasStride,
    half_t* __restrict__ outH, float* __restrict__ outF,
    const half_t* __restrict__ resH, const int* __restrict__ resmap,
    const float* __restrict__ lng, const float* __restrict__ lnb,
    half_t* __restrict__ out2H,
    const int* __restrict__ smap, float* __restrict__ gsum,
    const float* __restrict__ gate, const float* __restrict__ fin) {
    constexpr int LDA = KD + 8;
    __shared__ half_t As[32 * LDA];
    __shared__ float Cs[32][ND + 4];
    int tid = threadIdx.x;
    int m0 = blockIdx.x * 32;

    // stage A rows (optional gather / f32 conversion)
    for (int idx = tid; idx < 32 * (KD / 8); idx += 256) {
        int r = idx / (KD / 8);
        int kc = (idx - r * (KD / 8)) * 8;
        int src = rowmap ? rowmap[m0 + r] : (m0 + r);
        if (A32) {
            v8h t;
            const float* p = A32 + (size_t)src * KD + kc;
#pragma unroll
            for (int i = 0; i < 8; ++i) t[i] = (half_t)p[i];
            *(v8h*)&As[r * LDA + kc] = t;
        } else {
            *(v8h*)&As[r * LDA + kc] = *(const v8h*)(Ah + (size_t)src * KD + kc);
        }
    }
    __syncthreads();

    int wave = tid >> 5, lane = tid & 31;
    constexpr int NT = ND / 16;        // col tiles
    constexpr int TPW = (2 * NT) / 8;  // tiles per wave (all in one row tile)
    int t0 = wave * TPW;
    int rowTile = t0 / NT;
    int mA = rowTile * 16 + (lane & 15);
    int kb = (lane >> 4) * 8;

    v8f acc[TPW];
#pragma unroll
    for (int i = 0; i < TPW; ++i) acc[i] = v8f{0.f, 0.f, 0.f, 0.f, 0.f, 0.f, 0.f, 0.f};

    for (int k0 = 0; k0 < KD; k0 += 32) {
        v16h a = ldfrag(&As[mA * LDA + k0 + kb], &As[mA * LDA + k0 + 16 + kb]);
#pragma unroll
        for (int i = 0; i < TPW; ++i) {
            int ct = (t0 + i) % NT;
            int n = ct * 16 + (lane & 15);
            const half_t* wp = Wt + (size_t)n * KD + k0 + kb;
            v16h b = ldfrag(wp, wp + 16);
            acc[i] = wmma16(a, b, acc[i]);
        }
    }
#pragma unroll
    for (int i = 0; i < TPW; ++i) {
        int ct = (t0 + i) % NT;
        int col = ct * 16 + (lane & 15);
        int rbase = rowTile * 16 + ((lane >> 4) << 3);
#pragma unroll
        for (int r = 0; r < 8; ++r) Cs[rbase + r][col] = acc[i][r];
    }
    __syncthreads();

    if constexpr (MODE == 5) {
        // ND == 256 and blockDim == 256: thread owns column tid
        int n = tid;
        int cloud = m0 / CLOUD_L;
        float colsum = 0.f;
        for (int r = 0; r < 32; ++r) {
            int row = m0 + r;
            float v = Cs[r][n] + bias[n] + (float)resH[(size_t)row * ND + n];
            outH[(size_t)smap[row] * ND + n] = (half_t)v;
            colsum += v;
        }
        atomicAdd(&gsum[cloud * ND + n], colsum);
    } else {
        for (int e = tid; e < 32 * ND; e += 256) {
            int r = e / ND, n = e - r * ND;
            int row = m0 + r;
            float v = Cs[r][n];
            if (bias) {
                int bs = biasStride ? (row / CLOUD_L) * biasStride : 0;
                v += bias[bs + n];
            }
            if constexpr (MODE == 1) v = fmaxf(v, 0.f);
            if constexpr (MODE == 2) v = 0.5f * v * (1.f + erff(v * 0.70710678118f));
            if constexpr (MODE == 0 || MODE == 1 || MODE == 2) {
                outH[(size_t)row * ND + n] = (half_t)v;
            } else if constexpr (MODE == 3) {
                outF[(size_t)row * ND + n] = 1.f / (1.f + expf(-v));
            } else if constexpr (MODE == 4) {
                int rs = resmap ? resmap[row] : row;
                v += (float)resH[(size_t)rs * ND + n];
                outH[(size_t)row * ND + n] = (half_t)v;
                Cs[r][n] = v;
            } else if constexpr (MODE == 6) {
                size_t idx = (size_t)row * ND + n;
                outF[idx] = fin[idx] + gate[idx] * v;
            }
        }
    }
    if constexpr (MODE == 4) {
        __syncthreads();
        int wv = tid >> 5, ln = tid & 31;
        for (int rr = wv * 4; rr < wv * 4 + 4; ++rr) {
            float s = 0.f, s2 = 0.f;
            for (int j = ln; j < ND; j += 32) { float x = Cs[rr][j]; s += x; s2 += x * x; }
#pragma unroll
            for (int o = 16; o > 0; o >>= 1) { s += __shfl_xor(s, o); s2 += __shfl_xor(s2, o); }
            float mean = s * (1.f / ND);
            float inv = rsqrtf(s2 * (1.f / ND) - mean * mean + 1e-5f);
            for (int j = ln; j < ND; j += 32) {
                float x = (Cs[rr][j] - mean) * inv * lng[j] + lnb[j];
                out2H[(size_t)(m0 + rr) * ND + j] = (half_t)x;
            }
        }
    }
}

// ---------------------------------------------------------------------------
// Windowed attention: one block (16 waves) per (window, head).
// qkv serialized [N][768] f16 (Q|K|V), attn out [N][256] f16 serialized.
// ---------------------------------------------------------------------------
__global__ __launch_bounds__(512) void k_attn(const half_t* __restrict__ qkv,
                                              half_t* __restrict__ attn) {
    int head = blockIdx.x & 7;
    int win = blockIdx.x >> 3;
    int s0 = win * 256;
    __shared__ half_t Vt[32 * 264];    // V transposed: Vt[n][k]
    __shared__ half_t Pr[256 * 264];   // softmax probs, row-major
    int tid = threadIdx.x, lane = tid & 31, wave = tid >> 5;

    for (int idx = tid; idx < 8192; idx += 512) {
        int r = idx >> 5, c = idx & 31;
        Vt[c * 264 + r] = qkv[(size_t)(s0 + r) * 768 + 512 + head * 32 + c];
    }
    __syncthreads();

    int m0 = wave * 16;
    int mr = m0 + (lane & 15);
    int kb = (lane >> 4) * 8;
    const half_t* qrow = qkv + (size_t)(s0 + mr) * 768 + head * 32;
    v16h aq = ldfrag(qrow + kb, qrow + 16 + kb);

    v8f sc[16];
#pragma unroll
    for (int ct = 0; ct < 16; ++ct) {
        int n = ct * 16 + (lane & 15);
        const half_t* krow = qkv + (size_t)(s0 + n) * 768 + 256 + head * 32;
        v16h bk = ldfrag(krow + kb, krow + 16 + kb);
        v8f z = v8f{0.f, 0.f, 0.f, 0.f, 0.f, 0.f, 0.f, 0.f};
        sc[ct] = wmma16(aq, bk, z);
    }
    const float scale = 0.17677669529663687f;  // 1/sqrt(32)
#pragma unroll
    for (int r = 0; r < 8; ++r) {
        float mx = -1e30f;
#pragma unroll
        for (int ct = 0; ct < 16; ++ct) { sc[ct][r] *= scale; mx = fmaxf(mx, sc[ct][r]); }
#pragma unroll
        for (int o = 8; o > 0; o >>= 1) mx = fmaxf(mx, __shfl_xor(mx, o));
        float sum = 0.f;
#pragma unroll
        for (int ct = 0; ct < 16; ++ct) { float e = expf(sc[ct][r] - mx); sc[ct][r] = e; sum += e; }
#pragma unroll
        for (int o = 8; o > 0; o >>= 1) sum += __shfl_xor(sum, o);
        float inv = 1.f / sum;
        int prow = m0 + r + ((lane >> 4) << 3);
#pragma unroll
        for (int ct = 0; ct < 16; ++ct)
            Pr[prow * 264 + ct * 16 + (lane & 15)] = (half_t)(sc[ct][r] * inv);
    }
    __syncthreads();

    v8f o0 = v8f{0.f, 0.f, 0.f, 0.f, 0.f, 0.f, 0.f, 0.f};
    v8f o1 = o0;
    for (int k0 = 0; k0 < 256; k0 += 32) {
        v16h ap = ldfrag(&Pr[mr * 264 + k0 + kb], &Pr[mr * 264 + k0 + 16 + kb]);
        int n = (lane & 15);
        v16h b0 = ldfrag(&Vt[n * 264 + k0 + kb], &Vt[n * 264 + k0 + 16 + kb]);
        o0 = wmma16(ap, b0, o0);
        v16h b1 = ldfrag(&Vt[(16 + n) * 264 + k0 + kb], &Vt[(16 + n) * 264 + k0 + 16 + kb]);
        o1 = wmma16(ap, b1, o1);
    }
    int rbase = m0 + ((lane >> 4) << 3);
#pragma unroll
    for (int r = 0; r < 8; ++r) {
        size_t base = (size_t)(s0 + rbase + r) * 256 + head * 32;
        attn[base + (lane & 15)] = (half_t)o0[r];
        attn[base + 16 + (lane & 15)] = (half_t)o1[r];
    }
}

// ---------------------------------------------------------------------------
// Per-cloud: gt = gsum/L; ggt = gg_b1 + gt @ gg_w1[256:512]; fgt likewise.
// ---------------------------------------------------------------------------
__global__ __launch_bounds__(256) void k_cloudbias(
    const float* __restrict__ gsum,
    const float* __restrict__ gg_w1, const float* __restrict__ gg_b1,
    const float* __restrict__ f_w1, const float* __restrict__ f_b1,
    float* __restrict__ ggt, float* __restrict__ fgt) {
    int c = blockIdx.x, n = threadIdx.x;
    __shared__ float gsh[256];
    gsh[n] = gsum[c * 256 + n] * (1.f / (float)CLOUD_L);
    __syncthreads();
    float a = gg_b1[n], b = f_b1[n];
    for (int k = 0; k < 256; ++k) {
        float gk = gsh[k];
        a += gk * gg_w1[(size_t)(256 + k) * 256 + n];
        b += gk * f_w1[(size_t)(256 + k) * 256 + n];
    }
    ggt[c * 256 + n] = a;
    fgt[c * 256 + n] = b;
}

// ---------------------------------------------------------------------------
extern "C" void kernel_launch(void* const* d_in, const int* in_sizes, int n_in,
                              void* d_out, int out_size, void* d_ws, size_t ws_size,
                              hipStream_t stream) {
    const float* feats = (const float*)d_in[0];
    const float* coord = (const float*)d_in[1];
    const float* pe_w1 = (const float*)d_in[3];
    const float* pe_b1 = (const float*)d_in[4];
    const float* pe_w2 = (const float*)d_in[5];
    const float* pe_b2 = (const float*)d_in[6];
    const float* pre_g = (const float*)d_in[7];
    const float* pre_b = (const float*)d_in[8];
    const float* n1_g = (const float*)d_in[9];
    const float* n1_b = (const float*)d_in[10];
    const float* wqkv = (const float*)d_in[11];
    const float* bqkv = (const float*)d_in[12];
    const float* wo = (const float*)d_in[13];
    const float* bo = (const float*)d_in[14];
    const float* n2_g = (const float*)d_in[15];
    const float* n2_b = (const float*)d_in[16];
    const float* m_w1 = (const float*)d_in[17];
    const float* m_b1 = (const float*)d_in[18];
    const float* m_w2 = (const float*)d_in[19];
    const float* m_b2 = (const float*)d_in[20];
    const float* gg_w1 = (const float*)d_in[21];
    const float* gg_b1 = (const float*)d_in[22];
    const float* gg_w2 = (const float*)d_in[23];
    const float* gg_b2 = (const float*)d_in[24];
    const float* f_w1 = (const float*)d_in[25];
    const float* f_b1 = (const float*)d_in[26];
    const float* f_w2 = (const float*)d_in[27];
    const float* f_b2 = (const float*)d_in[28];
    float* out = (float*)d_out;

    char* ws = (char*)d_ws;
    size_t off = 0;
    auto alloc = [&](size_t bytes) -> char* {
        char* p = ws + off;
        off = (off + bytes + 255) & ~(size_t)255;
        return p;
    };
    half_t* wqkvT = (half_t*)alloc((size_t)768 * 256 * 2);
    half_t* woT   = (half_t*)alloc((size_t)256 * 256 * 2);
    half_t* m1T   = (half_t*)alloc((size_t)512 * 256 * 2);
    half_t* m2T   = (half_t*)alloc((size_t)256 * 512 * 2);
    half_t* gg1T  = (half_t*)alloc((size_t)256 * 256 * 2);
    half_t* gg2T  = (half_t*)alloc((size_t)256 * 256 * 2);
    half_t* fw1T  = (half_t*)alloc((size_t)256 * 256 * 2);
    half_t* fw2T  = (half_t*)alloc((size_t)256 * 256 * 2);
    float* stats  = (float*)alloc(NCLOUD * 12 * 4);
    float* keys   = (float*)alloc((size_t)NPTS * 4);
    float* kbuf   = (float*)alloc((size_t)NPTS * 4);
    int* ibuf     = (int*)alloc((size_t)NPTS * 4);
    int* order    = (int*)alloc((size_t)NPTS * 4);
    float* gsum   = (float*)alloc(NCLOUD * 256 * 4);
    float* ggt    = (float*)alloc(NCLOUD * 256 * 4);
    float* fgt    = (float*)alloc(NCLOUD * 256 * 4);
    half_t* x16   = (half_t*)alloc((size_t)NPTS * 256 * 2);
    half_t* h1    = (half_t*)alloc((size_t)NPTS * 256 * 2);  // reused as g1
    half_t* qkvb  = (half_t*)alloc((size_t)NPTS * 768 * 2);  // reused as t (mlp hidden)
    half_t* attnb = (half_t*)alloc((size_t)NPTS * 256 * 2);  // reused as xo
    half_t* xs2   = (half_t*)alloc((size_t)NPTS * 256 * 2);
    half_t* h2    = (half_t*)alloc((size_t)NPTS * 256 * 2);  // reused as f1
    float* gatef  = (float*)alloc((size_t)NPTS * 256 * 4);
    (void)ws_size; (void)in_sizes; (void)n_in; (void)out_size;

    // 1. weight transposes (f32 -> f16, [N][K])
    auto tl = [&](const float* W, half_t* Wt, int K, int N, int rowOff) {
        k_transpose<<<(K * N + 255) / 256, 256, 0, stream>>>(W, Wt, K, N, rowOff);
    };
    tl(wqkv, wqkvT, 256, 768, 0);
    tl(wo, woT, 256, 256, 0);
    tl(m_w1, m1T, 256, 512, 0);
    tl(m_w2, m2T, 512, 256, 0);
    tl(gg_w1, gg1T, 256, 256, 0);
    tl(gg_w2, gg2T, 256, 256, 0);
    tl(f_w1, fw1T, 256, 256, 0);
    tl(f_w2, fw2T, 256, 256, 0);

    // 2. coord stats, 3. per-point PE/LN/key, 4. per-cloud argsort
    k_reduce<<<NCLOUD, 1024, 0, stream>>>(coord, stats);
    k_point<<<NPTS / 8, 256, 0, stream>>>(feats, coord, stats, pe_w1, pe_b1, pe_w2,
                                          pe_b2, pre_g, pre_b, n1_g, n1_b, x16, h1, keys);
    k_sort<<<NCLOUD, 1024, 0, stream>>>(keys, kbuf, ibuf, order);

    const int G = NPTS / 32;
    // 5. qkv = gather(h1, order) @ Wqkv + bqkv  -> serialized [N][768] f16
    gemm_k<256, 768, 0><<<G, 256, 0, stream>>>(h1, nullptr, order, wqkvT, bqkv, 0,
                                               qkvb, nullptr, nullptr, nullptr, nullptr,
                                               nullptr, nullptr, nullptr, nullptr,
                                               nullptr, nullptr);
    // 6. windowed attention
    k_attn<<<(NPTS / 256) * 8, 512, 0, stream>>>(qkvb, attnb);
    // 7. xs2 = gather(x,order) + attn @ Wo + bo ; h2 = LN(xs2, n2)
    gemm_k<256, 256, 4><<<G, 256, 0, stream>>>(attnb, nullptr, nullptr, woT, bo, 0,
                                               xs2, nullptr, x16, order, n2_g, n2_b,
                                               h2, nullptr, nullptr, nullptr, nullptr);
    // 8. t = gelu(h2 @ m_w1 + m_b1)  (t aliases qkv buffer)
    half_t* tbuf = qkvb;
    gemm_k<256, 512, 2><<<G, 256, 0, stream>>>(h2, nullptr, nullptr, m1T, m_b1, 0,
                                               tbuf, nullptr, nullptr, nullptr, nullptr,
                                               nullptr, nullptr, nullptr, nullptr,
                                               nullptr, nullptr);
    // 9. xo[order[s]] = xs2 + t @ m_w2 + m_b2 ; gsum += per-cloud column sums
    hipMemsetAsync(gsum, 0, NCLOUD * 256 * 4, stream);
    half_t* xo = attnb;
    gemm_k<512, 256, 5><<<G, 256, 0, stream>>>(tbuf, nullptr, nullptr, m2T, m_b2, 0,
                                               xo, nullptr, xs2, nullptr, nullptr,
                                               nullptr, nullptr, order, gsum,
                                               nullptr, nullptr);
    // 10. per-cloud gate/fuse bias from gt = mean(xo)
    k_cloudbias<<<NCLOUD, 256, 0, stream>>>(gsum, gg_w1, gg_b1, f_w1, f_b1, ggt, fgt);
    // 11. g1 = relu(f @ gg_w1_top + ggt[cloud])
    half_t* g1 = h1;
    gemm_k<256, 256, 1><<<G, 256, 0, stream>>>(nullptr, feats, nullptr, gg1T, ggt, 256,
                                               g1, nullptr, nullptr, nullptr, nullptr,
                                               nullptr, nullptr, nullptr, nullptr,
                                               nullptr, nullptr);
    // 12. f1 = relu(xo @ f_w1_top + fgt[cloud])
    half_t* f1 = h2;
    gemm_k<256, 256, 1><<<G, 256, 0, stream>>>(xo, nullptr, nullptr, fw1T, fgt, 256,
                                               f1, nullptr, nullptr, nullptr, nullptr,
                                               nullptr, nullptr, nullptr, nullptr,
                                               nullptr, nullptr);
    // 13. gate = sigmoid(g1 @ gg_w2 + gg_b2)  (f32)
    gemm_k<256, 256, 3><<<G, 256, 0, stream>>>(g1, nullptr, nullptr, gg2T, gg_b2, 0,
                                               nullptr, gatef, nullptr, nullptr, nullptr,
                                               nullptr, nullptr, nullptr, nullptr,
                                               nullptr, nullptr);
    // 14. out = f + gate * (f1 @ f_w2 + f_b2)
    gemm_k<256, 256, 6><<<G, 256, 0, stream>>>(f1, nullptr, nullptr, fw2T, f_b2, 0,
                                               nullptr, out, nullptr, nullptr, nullptr,
                                               nullptr, nullptr, nullptr, nullptr,
                                               gatef, feats);
}
